// SharedSelfAttentionV1_54554674593974
// MI455X (gfx1250) — matrix operand order repaired and checked by reading, not compiled, not run on hardware
//
#include <hip/hip_runtime.h>
#include <hip/hip_bf16.h>

// ---------------------------------------------------------------------------
// Hyperbolic GAT forward for MI455X (gfx1250, wave32, WMMA, async-LDS).
// B matrices are kept in N x K layout (== the PyTorch-style weight itself),
// so both A and B WMMA fragments are contiguous 16-byte runs per lane.
// The big att GEMM stages A tiles into LDS with global_load_async_to_lds_b128
// (ASYNCcnt) double-buffered against the WMMA compute.
// ---------------------------------------------------------------------------

typedef __attribute__((ext_vector_type(16))) _Float16 v16h;
typedef __attribute__((ext_vector_type(8)))  _Float16 v8h;
typedef __attribute__((ext_vector_type(8)))  float    v8f;

#define NN    1024
#define DIN   128
#define DOUT  32
#define NE    32768

// ---------------- CDNA5 async-to-LDS helpers ----------------
__device__ __forceinline__ void async_ld_b128(unsigned lds_off, const void* gaddr) {
  asm volatile("global_load_async_to_lds_b128 %0, %1, off"
               :: "v"(lds_off), "v"(gaddr) : "memory");
}
__device__ __forceinline__ void wait_async0() {
  asm volatile("s_wait_asynccnt 0x0" ::: "memory");
}

// ---------------- wave32 helpers ----------------
__device__ __forceinline__ float wred(float v) {
  #pragma unroll
  for (int o = 16; o > 0; o >>= 1) v += __shfl_xor(v, o, 32);
  return v;
}

__device__ __forceinline__ float artanhf_(float x) {
  x = fminf(fmaxf(x, -1.f + 1e-7f), 1.f - 1e-7f);
  return 0.5f * (log1pf(x) - log1pf(-x));
}
__device__ __forceinline__ float arcoshf_(float x) {
  x = fmaxf(x, 1.f + 1e-7f);
  return logf(x + sqrtf(x * x - 1.f));
}

// lane j holds component j of a 32-dim hyperboloid vector (comp 0 = time).
__device__ __forceinline__ float expmap0_proj(float u, int lane) {
  float us = (lane == 0) ? 0.f : u;
  float xn = sqrtf(fmaxf(wred(us * us), 1e-15f));
  float y  = sinhf(xn) * us / xn;
  float x0 = sqrtf(fmaxf(1.f + wred(y * y), 1e-7f));
  return (lane == 0) ? x0 : y;
}

__device__ __forceinline__ float logmap0_w(float x, int lane) {
  float y  = (lane == 0) ? 0.f : x;
  float yn = sqrtf(fmaxf(wred(y * y), 1e-15f));
  float x0 = __shfl(x, 0, 32);
  float ac = arcoshf_(x0);
  return (lane == 0) ? 0.f : ac * y / yn;
}

__device__ __forceinline__ float mobius_bias(float xc, float bc, int lane) {
  float bt  = (lane == 0) ? 0.f : bc;
  float hb  = expmap0_proj(bt, lane);
  float lb  = logmap0_w(hb, lane);
  float x0  = __shfl(xc, 0, 32);
  float y   = (lane == 0) ? 0.f : xc;
  float yn  = sqrtf(fmaxf(wred(y * y), 1e-15f));
  float yu  = y / yn;
  float v   = (lane == 0) ? -yn : (1.f - x0) * yu;
  float lbs = (lane == 0) ? 0.f : lb;
  float alpha = wred(yu * lbs);
  float w   = lb - alpha * v;
  float ws  = (lane == 0) ? 0.f : w;
  float ux  = wred(y * ws);
  float u   = (lane == 0) ? ux / fmaxf(x0, 1e-7f) : w;
  float u0  = __shfl(u, 0, 32);
  float usq = wred(((lane == 0) ? 0.f : u) * u);
  float nu  = sqrtf(fmaxf(-u0 * u0 + usq, 1e-15f));
  float th  = fmaxf(nu, 1e-15f);
  float o   = coshf(th) * xc + sinhf(th) * u / th;
  float oy  = (lane == 0) ? 0.f : o;
  float o0  = sqrtf(fmaxf(1.f + wred(oy * oy), 1e-7f));
  return (lane == 0) ? o0 : oy;
}

// ---------------- WMMA fragment helpers ----------------
// A (16x32 f16): per lane two contiguous 8-half chunks at K = k+8*hi, k+16+8*hi
// B (32x16 f16) from BT (N x K): per lane two contiguous 8-half chunks at K = k+8*hi, k+16+8*hi
__device__ __forceinline__ v16h pack16(v8h lo, v8h hi) {
  v16h r;
  #pragma unroll
  for (int s = 0; s < 8; ++s) { r[s] = lo[s]; r[s + 8] = hi[s]; }
  return r;
}

// one wave -> one 16x16 C tile, direct global fragment loads
__global__ void __launch_bounds__(32)
wmma_gemm_direct(const _Float16* __restrict__ A, const _Float16* __restrict__ BT,
                 float* __restrict__ C, int K, int lda, int ldc) {
  const int lane = threadIdx.x & 31;
  const int hi = lane >> 4, l15 = lane & 15;
  const int m  = blockIdx.x * 16 + l15;
  const int n  = blockIdx.y * 16 + l15;

  v8f c = {0.f, 0.f, 0.f, 0.f, 0.f, 0.f, 0.f, 0.f};
  for (int k0 = 0; k0 < K; k0 += 32) {
    if (k0 + 32 < K)
      __builtin_prefetch(&A[(size_t)m * lda + k0 + 32], 0, 1);
    const _Float16* ap = A  + (size_t)m * lda + k0 + 8 * hi;
    const _Float16* bp = BT + (size_t)n * K  + k0 + 8 * hi;
    v16h a = pack16(*(const v8h*)ap, *(const v8h*)(ap + 16));
    v16h b = pack16(*(const v8h*)bp, *(const v8h*)(bp + 16));
    c = __builtin_amdgcn_wmma_f32_16x16x32_f16(false, a, false, b,
                                               (short)0, c, false, false);
  }
  #pragma unroll
  for (int r = 0; r < 8; ++r)
    C[(size_t)(blockIdx.x * 16 + r + 8 * hi) * ldc + n] = c[r];
}

// 8 waves -> 64x32 C block; A tiles staged via async-to-LDS, double buffered.
// N is fixed at 32 (ldc == 32).
__global__ void __launch_bounds__(256)
wmma_gemm_lds(const _Float16* __restrict__ A, const _Float16* __restrict__ BT,
              float* __restrict__ C, int K, int lda) {
  __shared__ __align__(16) _Float16 sA[2][64 * 32];
  const int t    = threadIdx.x;
  const int lane = t & 31, wv = t >> 5;
  const int hi = lane >> 4, l15 = lane & 15;
  const int tile_m = wv >> 1, tile_n = wv & 1;
  const int row0 = blockIdx.x * 64;
  const int srow = t >> 2, schunk = t & 3;      // 64 rows x 4 b128 chunks

  auto stage = [&](int k0, int buf) {
    const _Float16* g = A + (size_t)(row0 + srow) * lda + k0 + schunk * 8;
    unsigned l = (unsigned)(uintptr_t)&sA[buf][srow * 32 + schunk * 8];
    async_ld_b128(l, g);
  };

  stage(0, 0);
  const int n = tile_n * 16 + l15;
  v8f c = {0.f, 0.f, 0.f, 0.f, 0.f, 0.f, 0.f, 0.f};
  for (int k0 = 0, step = 0; k0 < K; k0 += 32, ++step) {
    const int cur = step & 1;
    wait_async0();          // my async stores to LDS landed
    __syncthreads();        // everyone's landed
    if (k0 + 32 < K) stage(k0 + 32, cur ^ 1);   // prefetch next tile
    const _Float16* abase = &sA[cur][(tile_m * 16 + l15) * 32];
    v16h a = pack16(*(const v8h*)(abase + 8 * hi),
                    *(const v8h*)(abase + 16 + 8 * hi));
    const _Float16* bp = BT + (size_t)n * K + k0 + 8 * hi;
    v16h b = pack16(*(const v8h*)bp, *(const v8h*)(bp + 16));
    c = __builtin_amdgcn_wmma_f32_16x16x32_f16(false, a, false, b,
                                               (short)0, c, false, false);
    __syncthreads();        // done reading cur before it is restaged
  }
  const int mrow = row0 + tile_m * 16;
  #pragma unroll
  for (int r = 0; r < 8; ++r)
    C[(size_t)(mrow + r + 8 * hi) * DOUT + n] = c[r];
}

// ---------------- stage kernels ----------------

__global__ void __launch_bounds__(128)
logmap_x_kernel(const float* __restrict__ x, _Float16* __restrict__ u16) {
  int row = blockIdx.x, t = threadIdx.x;
  float xv = x[(size_t)row * DIN + t];
  float y  = (t == 0) ? 0.f : xv;
  __shared__ float red[4];
  float s = wred(y * y);
  if ((t & 31) == 0) red[t >> 5] = s;
  __syncthreads();
  float yn = sqrtf(fmaxf(red[0] + red[1] + red[2] + red[3], 1e-15f));
  float x0 = x[(size_t)row * DIN];
  float r  = arcoshf_(x0) * y / yn;
  u16[(size_t)row * DIN + t] = (_Float16)((t == 0) ? 0.f : r);
}

// WT[r*kpad + k] = (k < kreal) ? W[r*kreal + k] : 0   (f16 convert + K pad)
__global__ void prep_w16(const float* __restrict__ W, _Float16* __restrict__ WT,
                         int n, int kpad, int kreal) {
  int i = blockIdx.x * blockDim.x + threadIdx.x;
  if (i >= n * kpad) return;
  int r = i / kpad, k = i % kpad;
  WT[i] = (_Float16)((k < kreal) ? W[r * kreal + k] : 0.f);
}

__global__ void zero_f16(_Float16* p, int n) {
  int i = blockIdx.x * blockDim.x + threadIdx.x;
  if (i < n) p[i] = (_Float16)0.f;
}

__global__ void __launch_bounds__(256)
emb_epilogue(const float* __restrict__ mm1, const float* __restrict__ bd,
             float* __restrict__ emb) {
  int lane = threadIdx.x & 31;
  int row  = blockIdx.x * 8 + (threadIdx.x >> 5);
  float res = expmap0_proj(mm1[(size_t)row * DOUT + lane], lane);
  emb[(size_t)row * DOUT + lane] = mobius_bias(res, bd[lane], lane);
}

__global__ void __launch_bounds__(256)
edge_feat(const float* __restrict__ emb, const int* __restrict__ edges,
          _Float16* __restrict__ ef) {
  int lane = threadIdx.x & 31;
  int e    = blockIdx.x * 8 + (threadIdx.x >> 5);
  int e0 = edges[e], e1 = edges[NE + e];
  float i0 = 1.f / (emb[(size_t)e0 * DOUT] + 1.f);
  float i1 = 1.f / (emb[(size_t)e1 * DOUT] + 1.f);
  auto cc = [&](int idx) -> float {
    if (idx < 31) return emb[(size_t)e0 * DOUT + idx + 1] * i0;
    if (idx < 62) return emb[(size_t)e1 * DOUT + idx - 30] * i1;
    return 0.f;
  };
  float cA = cc(lane), cB = cc(lane + 32);
  float sq = wred(cA * cA + cB * cB);
  float d  = fmaxf(1.f - sq, 1e-15f);
  float fp0 = (1.f + sq) / d;
  float fA = 2.f * cA / d, fB = 2.f * cB / d;
  float yn = sqrtf(fmaxf(wred(fA * fA + fB * fB), 1e-15f));
  float sc = arcoshf_(fp0) / yn;
  ef[(size_t)e * 64 + lane + 1] = (_Float16)(sc * fA);
  int   colB = (lane == 31) ? 0 : lane + 33;
  float vB   = (lane == 31) ? 0.f : sc * fB;
  ef[(size_t)e * 64 + colB] = (_Float16)vB;
}

__global__ void __launch_bounds__(256)
edge_epiA(const float* __restrict__ mm2, float* __restrict__ S) {
  int lane = threadIdx.x & 31;
  int e    = blockIdx.x * 8 + (threadIdx.x >> 5);
  float a1 = expmap0_proj(mm2[(size_t)e * DOUT + lane], lane);
  float u  = logmap0_w(a1, lane);
  u = (u > 0.f) ? u : 0.2f * u;
  if (lane == 0) u = 0.f;
  float a2 = expmap0_proj(u, lane);
  S[(size_t)e * DOUT + lane] = logmap0_w(a2, lane);
}

__global__ void __launch_bounds__(256)
col_softmax(float* __restrict__ S) {
  int col = blockIdx.x, t = threadIdx.x;
  __shared__ float sh[256];
  float m = -1e30f;
  for (int e = t; e < NE; e += 256) m = fmaxf(m, S[(size_t)e * DOUT + col]);
  sh[t] = m; __syncthreads();
  for (int o = 128; o > 0; o >>= 1) { if (t < o) sh[t] = fmaxf(sh[t], sh[t + o]); __syncthreads(); }
  m = sh[0]; __syncthreads();
  float s = 0.f;
  for (int e = t; e < NE; e += 256) s += __expf(S[(size_t)e * DOUT + col] - m);
  sh[t] = s; __syncthreads();
  for (int o = 128; o > 0; o >>= 1) { if (t < o) sh[t] += sh[t + o]; __syncthreads(); }
  float inv = 1.f / sh[0];
  for (int e = t; e < NE; e += 256)
    S[(size_t)e * DOUT + col] = __expf(S[(size_t)e * DOUT + col] - m) * inv;
}

__global__ void __launch_bounds__(256)
edge_att(const float* __restrict__ S, const float* __restrict__ emb,
         const int* __restrict__ edges, _Float16* __restrict__ att) {
  int lane = threadIdx.x & 31;
  int e    = blockIdx.x * 8 + (threadIdx.x >> 5);
  int e0 = edges[e], e1 = edges[NE + e];
  float u = S[(size_t)e * DOUT + lane];
  if (lane == 0) u = 0.f;
  float a  = expmap0_proj(u, lane);
  float a0 = __shfl(a, 0, 32);
  float pa = (lane == 0) ? 0.f : a / (a0 + 1.f);
  float w0 = emb[(size_t)e1 * DOUT];
  float pw = (lane == 0) ? 0.f : emb[(size_t)e1 * DOUT + lane] / (w0 + 1.f);
  float dot = wred(pa * pw);
  float wn  = sqrtf(fmaxf(wred(pw * pw), 1e-15f));
  float av  = tanhf(dot / wn * artanhf_(wn));
  if (lane == 0) att[(size_t)e0 * NN + e1] = (_Float16)av;
}

// B matrix for att GEMM, stored TRANSPOSED (32 x 1024) for fragment loads:
// row j (j<31) = lam*p_in[:,j], row 31 = lam-1
__global__ void __launch_bounds__(256)
bmat_kernel(const float* __restrict__ emb, _Float16* __restrict__ bmT) {
  int lane = threadIdx.x & 31;
  int i    = blockIdx.x * 8 + (threadIdx.x >> 5);
  float x  = emb[(size_t)i * DOUT + lane];
  float x0 = __shfl(x, 0, 32);
  float p  = (lane == 0) ? 0.f : x / (x0 + 1.f);
  float lam = 2.f / fmaxf(1.f - wred(p * p), 1e-15f);
  int   col = (lane == 0) ? 31 : lane - 1;
  float val = (lane == 0) ? (lam - 1.f) : lam * p;
  bmT[(size_t)col * NN + i] = (_Float16)val;
}

__global__ void __launch_bounds__(256)
final_kernel(const float* __restrict__ mm3, const float* __restrict__ Wo,
             const float* __restrict__ bo, float* __restrict__ out) {
  int lane = threadIdx.x & 31;
  int i    = blockIdx.x * 8 + (threadIdx.x >> 5);
  float den = mm3[(size_t)i * DOUT + 31];
  den = (fabsf(den) < 1e-10f) ? 1e-10f : den;
  float g  = (lane == 0) ? 0.f : mm3[(size_t)i * DOUT + lane - 1] / den;
  float gn = sqrtf(fmaxf(wred(g * g), 1e-15f));
  float po = tanhf(0.5f * artanhf_(gn)) * g / gn;
  float sq = wred(po * po);
  float dd = fmaxf(1.f - sq, 1e-15f);
  float ha = (lane == 0) ? (1.f + sq) / dd : 2.f * po / dd;
  float ul = logmap0_w(ha, lane);
  float acc = 0.f;
  #pragma unroll
  for (int d = 0; d < DOUT; ++d)
    acc += __shfl(ul, d, 32) * Wo[lane * DOUT + d];
  float res = expmap0_proj(acc, lane);
  out[(size_t)i * DOUT + lane] = mobius_bias(res, bo[lane], lane);
}

// ---------------------------------------------------------------------------
extern "C" void kernel_launch(void* const* d_in, const int* in_sizes, int n_in,
                              void* d_out, int out_size, void* d_ws, size_t ws_size,
                              hipStream_t stream) {
  const float* x     = (const float*)d_in[0];
  const int*   edges = (const int*)d_in[1];
  const float* Wd    = (const float*)d_in[2];
  const float* bd    = (const float*)d_in[3];
  const float* Wa    = (const float*)d_in[4];
  const float* Wo    = (const float*)d_in[5];
  const float* bo    = (const float*)d_in[6];
  float* out = (float*)d_out;

  char* ws = (char*)d_ws;
  size_t off = 0;
  auto carve = [&](size_t bytes) { void* p = ws + off; off = (off + bytes + 255) & ~(size_t)255; return p; };
  _Float16* u16    = (_Float16*)carve((size_t)NN * DIN * 2);
  _Float16* Wd16   = (_Float16*)carve((size_t)DOUT * DIN * 2);   // N x K
  float*    mm1    = (float*)   carve((size_t)NN * DOUT * 4);
  float*    emb    = (float*)   carve((size_t)NN * DOUT * 4);
  _Float16* ef16   = (_Float16*)carve((size_t)NE * 64 * 2);
  _Float16* Wa16   = (_Float16*)carve((size_t)DOUT * 64 * 2);    // N x Kpad
  float*    mm2    = (float*)   carve((size_t)NE * DOUT * 4);
  float*    Sbuf   = (float*)   carve((size_t)NE * DOUT * 4);
  _Float16* att16  = (_Float16*)carve((size_t)NN * NN * 2);
  _Float16* bmT16  = (_Float16*)carve((size_t)DOUT * NN * 2);    // N x K
  float*    mm3    = (float*)   carve((size_t)NN * DOUT * 4);
  (void)ws_size; (void)in_sizes; (void)n_in; (void)out_size;

  logmap_x_kernel<<<NN, 128, 0, stream>>>(x, u16);
  prep_w16<<<(DOUT * DIN + 255) / 256, 256, 0, stream>>>(Wd, Wd16, DOUT, DIN, DIN);
  prep_w16<<<(DOUT * 64 + 255) / 256, 256, 0, stream>>>(Wa, Wa16, DOUT, 64, 63);

  wmma_gemm_direct<<<dim3(NN / 16, DOUT / 16), 32, 0, stream>>>(u16, Wd16, mm1, DIN, DIN, DOUT);
  emb_epilogue<<<NN / 8, 256, 0, stream>>>(mm1, bd, emb);

  edge_feat<<<NE / 8, 256, 0, stream>>>(emb, edges, ef16);
  wmma_gemm_direct<<<dim3(NE / 16, DOUT / 16), 32, 0, stream>>>(ef16, Wa16, mm2, 64, 64, DOUT);
  edge_epiA<<<NE / 8, 256, 0, stream>>>(mm2, Sbuf);
  col_softmax<<<DOUT, 256, 0, stream>>>(Sbuf);

  zero_f16<<<(NN * NN + 255) / 256, 256, 0, stream>>>(att16, NN * NN);
  edge_att<<<NE / 8, 256, 0, stream>>>(Sbuf, emb, edges, att16);
  bmat_kernel<<<NN / 8, 256, 0, stream>>>(emb, bmT16);

  wmma_gemm_lds<<<NN / 64, 256, 0, stream>>>(att16, bmT16, mm3, NN, NN);
  final_kernel<<<NN / 8, 256, 0, stream>>>(mm3, Wo, bo, out);
}
